// MultiheadDiffAttn_12489764897544
// MI455X (gfx1250) — matrix operand order repaired and checked by reading, not compiled, not run on hardware
//
#include <hip/hip_runtime.h>
#include <hip/hip_bf16.h>

// ---------------------------------------------------------------------------
// Problem constants (from reference)
// ---------------------------------------------------------------------------
#define BB 2
#define TT 4096
#define DD 1024
#define HDIM 512
#define LAMBDA_INIT_F 0.7836057665316244f
#define ONE_MINUS_LI 0.2163942334683756f
#define SCALING_F 0.04419417382415922f
#define EPS_F 1e-5f

typedef __attribute__((ext_vector_type(16))) __bf16 v16bf;
typedef __attribute__((ext_vector_type(8)))  __bf16 v8bf;
typedef __attribute__((ext_vector_type(8)))  float  v8f;

// ---------------------------------------------------------------------------
// WMMA helpers. CDNA5 16x16x32 bf16, f32 accumulate.
// A fragment (16x32, M x K): lane = (m = lane&15, hi = lane>>4).
//   element i -> K = k0 + 16*hi + i            (contiguous 16 halfs)
// B fragment (32x16, K x N): lane = (n = lane&15, hi = lane>>4).
//   element i -> K = k0 + 16*hi + i            (contiguous 16 halfs)
// C/D (16x16 f32): VGPR r -> row m0 + r + 8*hi, col n0 + (lane&15).
// ---------------------------------------------------------------------------
__device__ __forceinline__ v8f wmma_bf16(v16bf a, v16bf b, v8f c) {
  return __builtin_amdgcn_wmma_f32_16x16x32_bf16(
      /*neg_a=*/false, a, /*neg_b=*/false, b,
      /*c_mod=*/(short)0, c, /*reuse_a=*/false, /*reuse_b=*/false);
}

__device__ __forceinline__ v16bf load_a_frag(const __bf16* __restrict__ row, int hi) {
  v8bf lo = *(const v8bf*)(row + 16 * hi);
  v8bf hh = *(const v8bf*)(row + 16 * hi + 8);
  return __builtin_shufflevector(lo, hh, 0, 1, 2, 3, 4, 5, 6, 7,
                                 8, 9, 10, 11, 12, 13, 14, 15);
}

// p = &B[n][k0] (K-major per column n); 32B-aligned by construction.
__device__ __forceinline__ v16bf load_b_frag(const __bf16* __restrict__ p, int hi) {
  return *(const v16bf*)(p + 16 * hi);
}

// ---------------------------------------------------------------------------
// f32 -> bf16 convert
// ---------------------------------------------------------------------------
__global__ void cvt_kernel(const float* __restrict__ x, __bf16* __restrict__ y, size_t n) {
  size_t i = (size_t)blockIdx.x * blockDim.x + threadIdx.x;
  size_t stride = (size_t)gridDim.x * blockDim.x;
  for (; i < n; i += stride) y[i] = (__bf16)x[i];
}

// ---------------------------------------------------------------------------
// lambda = exp(sum(lq1*lk1)) - exp(sum(lq2*lk2)) + LAMBDA_INIT
// ---------------------------------------------------------------------------
__global__ void lam_kernel(const float* __restrict__ q1, const float* __restrict__ k1,
                           const float* __restrict__ q2, const float* __restrict__ k2,
                           float* __restrict__ lam) {
  float s1 = 0.f, s2 = 0.f;
  for (int i = threadIdx.x; i < HDIM; i += blockDim.x) {
    s1 += q1[i] * k1[i];
    s2 += q2[i] * k2[i];
  }
#pragma unroll
  for (int m = 16; m >= 1; m >>= 1) { s1 += __shfl_xor(s1, m); s2 += __shfl_xor(s2, m); }
  __shared__ float r1[8], r2[8];
  int w = threadIdx.x >> 5;
  if ((threadIdx.x & 31) == 0) { r1[w] = s1; r2[w] = s2; }
  __syncthreads();
  if (threadIdx.x == 0) {
    float a = 0.f, b = 0.f;
#pragma unroll
    for (int i = 0; i < 8; ++i) { a += r1[i]; b += r2[i]; }
    lam[0] = __expf(a) - __expf(b) + LAMBDA_INIT_F;
  }
}

// ---------------------------------------------------------------------------
// Generic bf16 WMMA GEMM:  C[M,N] = alpha * A[M,K] x B   (B given K-major per n:
// element B[k][n] at Bm[n*ldb + k], which covers both x*W^T and vpT layouts).
// Each wave computes a 16x128 tile (8 accumulators). blockDim = 256 (8 waves).
// Four independent B-fragment buffers, consume-then-refill: each WMMA operand
// was loaded 4 fragments earlier.
// MODE 0: bf16 normal store; MODE 1: bf16 transposed store (C[n][m], ldc = col
// stride along m); MODE 2: f32 normal store.
// ---------------------------------------------------------------------------
template <int MODE>
__global__ __launch_bounds__(256, 1) void gemm_k(
    const __bf16* __restrict__ A, int lda, long long sA,
    const __bf16* __restrict__ Bm, int ldb, long long sB,
    void* __restrict__ Cv, int ldc, long long sC,
    int M, int N, int K, float alpha) {
  int wave = (int)((blockIdx.x * blockDim.x + threadIdx.x) >> 5);
  int lane = threadIdx.x & 31;
  int nlo = lane & 15, hi = lane >> 4;
  int nT = N >> 7;  // 128-wide groups
  int m0 = (wave / nT) << 4;
  int n0 = (wave % nT) << 7;
  if (m0 >= M) return;
  int z = blockIdx.z;
  A += (size_t)z * (size_t)sA;
  Bm += (size_t)z * (size_t)sB;

  const __bf16* Arow = A + (size_t)(m0 + nlo) * (size_t)lda;
  const __bf16* Bcol = Bm + (size_t)(n0 + nlo) * (size_t)ldb;
  const size_t ldb16 = (size_t)16 * (size_t)ldb;
  v8f acc[8] = {};
  for (int k0 = 0; k0 < K; k0 += 32) {
    __builtin_prefetch(Arow + k0 + 256, 0, 0);
    v16bf a = load_a_frag(Arow + k0, hi);
    v16bf f0 = load_b_frag(Bcol + 0 * ldb16 + k0, hi);
    v16bf f1 = load_b_frag(Bcol + 1 * ldb16 + k0, hi);
    v16bf f2 = load_b_frag(Bcol + 2 * ldb16 + k0, hi);
    v16bf f3 = load_b_frag(Bcol + 3 * ldb16 + k0, hi);
    acc[0] = wmma_bf16(a, f0, acc[0]);
    f0 = load_b_frag(Bcol + 4 * ldb16 + k0, hi);
    acc[1] = wmma_bf16(a, f1, acc[1]);
    f1 = load_b_frag(Bcol + 5 * ldb16 + k0, hi);
    acc[2] = wmma_bf16(a, f2, acc[2]);
    f2 = load_b_frag(Bcol + 6 * ldb16 + k0, hi);
    acc[3] = wmma_bf16(a, f3, acc[3]);
    f3 = load_b_frag(Bcol + 7 * ldb16 + k0, hi);
    acc[4] = wmma_bf16(a, f0, acc[4]);
    acc[5] = wmma_bf16(a, f1, acc[5]);
    acc[6] = wmma_bf16(a, f2, acc[6]);
    acc[7] = wmma_bf16(a, f3, acc[7]);
  }

  if (MODE == 0) {
    __bf16* C = (__bf16*)Cv + (size_t)z * (size_t)sC;
#pragma unroll
    for (int j = 0; j < 8; ++j)
#pragma unroll
      for (int r = 0; r < 8; ++r)
        C[(size_t)(m0 + r + 8 * hi) * (size_t)ldc + n0 + j * 16 + nlo] =
            (__bf16)(acc[j][r] * alpha);
  } else if (MODE == 1) {
    __bf16* C = (__bf16*)Cv + (size_t)z * (size_t)sC;
#pragma unroll
    for (int j = 0; j < 8; ++j) {
      v8bf pk;
#pragma unroll
      for (int r = 0; r < 8; ++r) pk[r] = (__bf16)(acc[j][r] * alpha);
      *(v8bf*)(C + (size_t)(n0 + j * 16 + nlo) * (size_t)ldc + m0 + 8 * hi) = pk;
    }
  } else {
    float* C = (float*)Cv + (size_t)z * (size_t)sC;
#pragma unroll
    for (int j = 0; j < 8; ++j)
#pragma unroll
      for (int r = 0; r < 8; ++r)
        C[(size_t)(m0 + r + 8 * hi) * (size_t)ldc + n0 + j * 16 + nlo] = acc[j][r] * alpha;
  }
}

// ---------------------------------------------------------------------------
// Score sweep core: 16 resident Q fragments x streaming K fragments.
// Four independent B buffers, consume-then-refill, unroll by 4: each WMMA's
// operand was issued 4 fragments (8 b128 loads) earlier.
// ---------------------------------------------------------------------------
__device__ __forceinline__ v8f score_tile(const v16bf* qa, const __bf16* __restrict__ Bb,
                                          int hi) {
  v8f c = {};
  v16bf f0 = load_b_frag(Bb + 0 * 32, hi);
  v16bf f1 = load_b_frag(Bb + 1 * 32, hi);
  v16bf f2 = load_b_frag(Bb + 2 * 32, hi);
  v16bf f3 = load_b_frag(Bb + 3 * 32, hi);
#pragma unroll
  for (int kk = 0; kk < 16; kk += 4) {
    c = wmma_bf16(qa[kk + 0], f0, c);
    if (kk + 4 < 16) f0 = load_b_frag(Bb + (kk + 4) * 32, hi);
    c = wmma_bf16(qa[kk + 1], f1, c);
    if (kk + 5 < 16) f1 = load_b_frag(Bb + (kk + 5) * 32, hi);
    c = wmma_bf16(qa[kk + 2], f2, c);
    if (kk + 6 < 16) f2 = load_b_frag(Bb + (kk + 6) * 32, hi);
    c = wmma_bf16(qa[kk + 3], f3, c);
    if (kk + 7 < 16) f3 = load_b_frag(Bb + (kk + 7) * 32, hi);
  }
  return c;
}

// ---------------------------------------------------------------------------
// Softmax stats pass: per (b, head, 16-row tile), sweep s in 16-wide tiles,
// online row max + sum(exp). Row reduction = shfl_xor over the 16-lane half.
// smax/ssum layout: [(b*2+h)*T + t].
// ---------------------------------------------------------------------------
__global__ __launch_bounds__(256, 1) void stats_k(
    const __bf16* __restrict__ qp, const __bf16* __restrict__ kp,
    float* __restrict__ smax, float* __restrict__ ssum) {
  int wave = (int)((blockIdx.x * blockDim.x + threadIdx.x) >> 5);
  int lane = threadIdx.x & 31;
  int nlo = lane & 15, hi = lane >> 4;
  const int tilesPerSeq = TT / 16;
  int t0 = (wave % tilesPerSeq) << 4;
  int bh = wave / tilesPerSeq;  // b*2 + h
  int h = bh & 1, b = bh >> 1;

  const __bf16* Arow = qp + (size_t)(b * TT + t0 + nlo) * DD + h * HDIM;
  v16bf qa[16];
#pragma unroll
  for (int kk = 0; kk < 16; ++kk) qa[kk] = load_a_frag(Arow + kk * 32, hi);

  float rm[8], rs[8];
#pragma unroll
  for (int r = 0; r < 8; ++r) { rm[r] = -3.0e38f; rs[r] = 0.f; }

  for (int s0 = 0; s0 < TT; s0 += 16) {
    const __bf16* Bb = kp + (size_t)(b * TT + s0 + nlo) * HDIM;
    v8f c = score_tile(qa, Bb, hi);
#pragma unroll
    for (int r = 0; r < 8; ++r) {
      float x = c[r];
      float mx = x;
      mx = fmaxf(mx, __shfl_xor(mx, 1));
      mx = fmaxf(mx, __shfl_xor(mx, 2));
      mx = fmaxf(mx, __shfl_xor(mx, 4));
      mx = fmaxf(mx, __shfl_xor(mx, 8));
      float mn = fmaxf(rm[r], mx);
      float e = __expf(x - mn);
      e += __shfl_xor(e, 1);
      e += __shfl_xor(e, 2);
      e += __shfl_xor(e, 4);
      e += __shfl_xor(e, 8);
      rs[r] = rs[r] * __expf(rm[r] - mn) + e;
      rm[r] = mn;
    }
  }
  if (nlo == 0) {
#pragma unroll
    for (int r = 0; r < 8; ++r) {
      size_t idx = (size_t)bh * TT + t0 + r + 8 * hi;
      smax[idx] = rm[r];
      ssum[idx] = rs[r];
    }
  }
}

// ---------------------------------------------------------------------------
// Weight pass: recompute scores for head h, W[b,t,s] (bf16).
// h==0: W = exp(s0 - m0)/sum0.   h==1: W -= lam * exp(s1 - m1)/sum1.
// ---------------------------------------------------------------------------
__global__ __launch_bounds__(256, 1) void weights_k(
    const __bf16* __restrict__ qp, const __bf16* __restrict__ kp,
    const float* __restrict__ smax, const float* __restrict__ ssum,
    const float* __restrict__ lamp, __bf16* __restrict__ W, int h) {
  int wave = (int)((blockIdx.x * blockDim.x + threadIdx.x) >> 5);
  int lane = threadIdx.x & 31;
  int nlo = lane & 15, hi = lane >> 4;
  const int tilesPerSeq = TT / 16;
  int t0 = (wave % tilesPerSeq) << 4;
  int b = wave / tilesPerSeq;

  const __bf16* Arow = qp + (size_t)(b * TT + t0 + nlo) * DD + h * HDIM;
  v16bf qa[16];
#pragma unroll
  for (int kk = 0; kk < 16; ++kk) qa[kk] = load_a_frag(Arow + kk * 32, hi);

  float mrow[8], isr[8];
#pragma unroll
  for (int r = 0; r < 8; ++r) {
    size_t idx = (size_t)(b * 2 + h) * TT + t0 + r + 8 * hi;
    mrow[r] = smax[idx];
    isr[r] = 1.f / ssum[idx];
  }
  float lam = lamp[0];

  for (int s0 = 0; s0 < TT; s0 += 16) {
    const __bf16* Bb = kp + (size_t)(b * TT + s0 + nlo) * HDIM;
    v8f c = score_tile(qa, Bb, hi);
#pragma unroll
    for (int r = 0; r < 8; ++r) {
      float val = __expf(c[r] - mrow[r]) * isr[r];
      __bf16* wp = W + (size_t)(b * TT + t0 + r + 8 * hi) * TT + s0 + nlo;
      if (h == 0)
        *wp = (__bf16)val;
      else
        *wp = (__bf16)((float)*wp - lam * val);
    }
  }
}

// ---------------------------------------------------------------------------
// RMSNorm over D=1024 (no affine), scale by (1 - LAMBDA_INIT), emit bf16.
// ---------------------------------------------------------------------------
__global__ void rms_kernel(const float* __restrict__ X, __bf16* __restrict__ Y) {
  int row = blockIdx.x;
  const float* x = X + (size_t)row * DD;
  float ss = 0.f;
  for (int i = threadIdx.x; i < DD; i += blockDim.x) { float v = x[i]; ss += v * v; }
#pragma unroll
  for (int m = 16; m >= 1; m >>= 1) ss += __shfl_xor(ss, m);
  __shared__ float red[8];
  int w = threadIdx.x >> 5;
  if ((threadIdx.x & 31) == 0) red[w] = ss;
  __syncthreads();
  if (threadIdx.x < 8) {
    float s = red[threadIdx.x];
    s += __shfl_xor(s, 4);
    s += __shfl_xor(s, 2);
    s += __shfl_xor(s, 1);
    if (threadIdx.x == 0) red[0] = s;
  }
  __syncthreads();
  float scale = rsqrtf(red[0] * (1.0f / DD) + EPS_F) * ONE_MINUS_LI;
  for (int i = threadIdx.x; i < DD; i += blockDim.x)
    Y[(size_t)row * DD + i] = (__bf16)(x[i] * scale);
}

// ---------------------------------------------------------------------------
// Host-side launch
// ---------------------------------------------------------------------------
extern "C" void kernel_launch(void* const* d_in, const int* in_sizes, int n_in,
                              void* d_out, int out_size, void* d_ws, size_t ws_size,
                              hipStream_t stream) {
  (void)in_sizes; (void)n_in; (void)out_size; (void)ws_size;
  const float* q   = (const float*)d_in[0];
  const float* k   = (const float*)d_in[1];
  const float* v   = (const float*)d_in[2];
  const float* Wq  = (const float*)d_in[3];
  const float* Wk  = (const float*)d_in[4];
  const float* Wv  = (const float*)d_in[5];
  const float* Wo  = (const float*)d_in[6];
  const float* lq1 = (const float*)d_in[7];
  const float* lk1 = (const float*)d_in[8];
  const float* lq2 = (const float*)d_in[9];
  const float* lk2 = (const float*)d_in[10];
  float* out = (float*)d_out;

  const size_t BT = (size_t)BB * TT;  // 8192

  // --- workspace layout (256B aligned slots) ---
  char* ws = (char*)d_ws;
  size_t off = 0;
  auto alloc = [&](size_t bytes) -> char* {
    char* p = ws + off;
    off += (bytes + 255) & ~(size_t)255;
    return p;
  };
  __bf16* q_bf   = (__bf16*)alloc(BT * DD * 2);
  __bf16* k_bf   = (__bf16*)alloc(BT * DD * 2);
  __bf16* v_bf   = (__bf16*)alloc(BT * DD * 2);
  __bf16* wq_bf  = (__bf16*)alloc((size_t)DD * DD * 2);
  __bf16* wk_bf  = (__bf16*)alloc((size_t)HDIM * DD * 2);
  __bf16* wv_bf  = (__bf16*)alloc((size_t)DD * DD * 2);
  __bf16* wo_bf  = (__bf16*)alloc((size_t)DD * DD * 2);
  __bf16* qp_bf  = (__bf16*)alloc(BT * DD * 2);            // [b*T+t][1024], heads in cols
  __bf16* kp_bf  = (__bf16*)alloc(BT * HDIM * 2);          // [b*T+s][512]
  __bf16* vpT_bf = (__bf16*)alloc((size_t)DD * BT * 2);    // [d][b*T+s]
  float*  smax   = (float*)alloc((size_t)BB * 2 * TT * 4);
  float*  ssum   = (float*)alloc((size_t)BB * 2 * TT * 4);
  float*  lamp   = (float*)alloc(256);
  __bf16* Wt     = (__bf16*)alloc(BT * TT * 2);            // 67 MB, L2-resident
  float*  attn_f = (float*)alloc(BT * DD * 4);
  __bf16* attn_b = (__bf16*)alloc(BT * DD * 2);

  // --- 1. convert inputs to bf16 ---
  auto cvt = [&](const float* src, __bf16* dst, size_t n) {
    int blocks = (int)((n + 1023) / 1024);
    cvt_kernel<<<blocks, 256, 0, stream>>>(src, dst, n);
  };
  cvt(q, q_bf, BT * DD);
  cvt(k, k_bf, BT * DD);
  cvt(v, v_bf, BT * DD);
  cvt(Wq, wq_bf, (size_t)DD * DD);
  cvt(Wk, wk_bf, (size_t)HDIM * DD);
  cvt(Wv, wv_bf, (size_t)DD * DD);
  cvt(Wo, wo_bf, (size_t)DD * DD);

  // --- 2. lambda scalar ---
  lam_kernel<<<1, 256, 0, stream>>>(lq1, lk1, lq2, lk2, lamp);

  // --- 3. projections ---
  // qp = SCALING * q @ Wq^T  : [8192,1024], waves = 512*8
  gemm_k<0><<<dim3(512, 1, 1), 256, 0, stream>>>(
      q_bf, DD, 0, wq_bf, DD, 0, qp_bf, DD, 0, (int)BT, DD, DD, SCALING_F);
  // kp = k @ Wk^T : [8192,512], waves = 512*4
  gemm_k<0><<<dim3(256, 1, 1), 256, 0, stream>>>(
      k_bf, DD, 0, wk_bf, DD, 0, kp_bf, HDIM, 0, (int)BT, HDIM, DD, 1.0f);
  // vpT[d][b*T+s] = (v @ Wv^T)^T : transposed bf16 store, col-stride BT
  gemm_k<1><<<dim3(512, 1, 1), 256, 0, stream>>>(
      v_bf, DD, 0, wv_bf, DD, 0, vpT_bf, (int)BT, 0, (int)BT, DD, DD, 1.0f);

  // --- 4. softmax stats (both heads): 1024 waves ---
  stats_k<<<dim3(128, 1, 1), 256, 0, stream>>>(qp_bf, kp_bf, smax, ssum);

  // --- 5. combined weights W = p1 - lam*p2 (two passes, RMW): 512 waves each ---
  weights_k<<<dim3(64, 1, 1), 256, 0, stream>>>(qp_bf, kp_bf, smax, ssum, lamp, Wt, 0);
  weights_k<<<dim3(64, 1, 1), 256, 0, stream>>>(qp_bf, kp_bf, smax, ssum, lamp, Wt, 1);

  // --- 6. attn = W @ vp : batched over b (A stride T*T, B col base stride T) ---
  gemm_k<2><<<dim3(256, 1, BB), 256, 0, stream>>>(
      Wt, TT, (long long)TT * TT,
      vpT_bf, (int)BT, (long long)TT,
      attn_f, DD, (long long)TT * DD,
      TT, DD, TT, 1.0f);

  // --- 7. RMSNorm + (1-lambda_init), emit bf16 ---
  rms_kernel<<<dim3((unsigned)BT, 1, 1), 256, 0, stream>>>(attn_f, attn_b);

  // --- 8. out = attn @ Wo^T (f32 to d_out) ---
  gemm_k<2><<<dim3(512, 1, 1), 256, 0, stream>>>(
      attn_b, DD, 0, wo_bf, DD, 0, out, DD, 0, (int)BT, DD, DD, 1.0f);
}
